// PredictorSiameseGED_25898652795264
// MI455X (gfx1250) — compile-verified, matrix operand order checked by reading, not run on hardware
//
#include <hip/hip_runtime.h>
#include <hip/hip_bf16.h>

// ---------------------------------------------------------------------------
// Siamese GIN forward for gfx1250 (MI455X).
// f32 data path (memory bound, L2-resident edge phase), f16-in/f32-acc WMMA
// (v_wmma_f32_16x16x32_f16) for the dense 32x32 MLP matmuls.
// ---------------------------------------------------------------------------

typedef __attribute__((ext_vector_type(16))) _Float16 v16h;
typedef __attribute__((ext_vector_type(8)))  float    v8f;

constexpr int N  = 524288;
constexpr int E  = 2097152;
constexpr int G  = 65536;        // N / 8 graphs
constexpr int NF = N * 32;       // padded feature elements

#define ATOM_ADD(p, v) __hip_atomic_fetch_add((p), (v), __ATOMIC_RELAXED, __HIP_MEMORY_SCOPE_AGENT)

// K index held by element i (0..15) of a lane in half h (lane>>4) of an
// f16 16x32 A fragment (and symmetrically of a 32x16 B fragment).
__device__ __forceinline__ int kmap(int i, int h) {
    return (i < 8) ? (h * 8 + i) : (16 + h * 8 + (i - 8));
}

// Build a B fragment for output columns [n0, n0+16) of weight W[fan_in, fan_out]
// (row-major, fan_out = 32), zero-padding K >= fan_in.
__device__ __forceinline__ v16h make_bfrag(const float* __restrict__ W,
                                           int fan_in, int n0, int lane) {
    const int n = n0 + (lane & 15);
    const int h = lane >> 4;
    v16h b;
#pragma unroll
    for (int i = 0; i < 16; ++i) {
        const int K = kmap(i, h);
        b[i] = (_Float16)((K < fan_in) ? W[K * 32 + n] : 0.0f);
    }
    return b;
}

// ----------------------------- small kernels -------------------------------

__global__ void k_zero(float* __restrict__ agg, float* __restrict__ stats) {
    const int idx = blockIdx.x * 256 + threadIdx.x;
    if (idx < NF / 4) ((float4*)agg)[idx] = make_float4(0.f, 0.f, 0.f, 0.f);
    if (idx < 64) stats[idx] = 0.f;
}

__global__ void k_pack(const float* __restrict__ in, float* __restrict__ X) {
    const int idx = blockIdx.x * 256 + threadIdx.x;   // < NF
    const int n = idx >> 5, c = idx & 31;
    X[idx] = (c < 6) ? in[n * 6 + c] : 0.f;
}

__global__ void k_agg(const int* __restrict__ ei, const float* __restrict__ X,
                      float* __restrict__ agg, int D) {
    const int e = blockIdx.x * 256 + threadIdx.x;
    if (e >= E) return;
    const int s = ei[e];
    const int d = ei[E + e];
    const float* xs = X + (long)s * 32;
    float* ad = agg + (long)d * 32;
    for (int c = 0; c < D; c += 4) {
        const float4 v = *(const float4*)(xs + c);
        ATOM_ADD(ad + c + 0, v.x);
        ATOM_ADD(ad + c + 1, v.y);
        ATOM_ADD(ad + c + 2, v.z);
        ATOM_ADD(ad + c + 3, v.w);
    }
}

__global__ void k_bn(const float* __restrict__ Z, const float* __restrict__ stats,
                     const float* __restrict__ gamma, const float* __restrict__ beta,
                     float* __restrict__ X) {
    const int idx = blockIdx.x * 256 + threadIdx.x;   // < NF
    const int c = idx & 31;
    const float inv_n = 1.0f / (float)N;
    const float mu  = stats[c] * inv_n;
    const float var = stats[32 + c] * inv_n - mu * mu;
    const float sc  = gamma[c] * rsqrtf(var + 1e-5f);
    X[idx] = (Z[idx] - mu) * sc + beta[c];
}

// ------------------------- GIN MLP (WMMA core) -----------------------------
// One wave = one 16-node tile. h = x + agg -> relu(h@W1+b1) -> @W2+b2 -> relu,
// z stored to global + per-channel sum / sum-of-squares atomics for BN.
__global__ void __launch_bounds__(256) k_mlp(
    const float* __restrict__ X,  const float* __restrict__ AGG,
    const float* __restrict__ W1, const float* __restrict__ B1,
    const float* __restrict__ W2, const float* __restrict__ B2,
    int fan_in, float* __restrict__ Z, float* __restrict__ stats) {

    __shared__ float lds[8][16 * 33];
    const int lane = threadIdx.x & 31;
    const int wave = threadIdx.x >> 5;
    float* Lp = &lds[wave][0];
    const int node0 = (blockIdx.x * 8 + wave) * 16;
    const int mr = lane & 15;
    const int h  = lane >> 4;
    const int mb = h * 8;              // C/D fragment row base for this lane half

    // Weight fragments (tiny; L0/L2 resident)
    const v16h bw1a = make_bfrag(W1, fan_in, 0, lane);
    const v16h bw1b = make_bfrag(W1, fan_in, 16, lane);
    const v16h bw2a = make_bfrag(W2, 32, 0, lane);
    const v16h bw2b = make_bfrag(W2, 32, 16, lane);

    // A fragment: h = x + agg for row (node0+mr), K-set of this lane half
    const float4* xp = (const float4*)(X   + (long)(node0 + mr) * 32 + h * 8);
    const float4* gp = (const float4*)(AGG + (long)(node0 + mr) * 32 + h * 8);
    const float4 x0 = xp[0], x1 = xp[1], x4 = xp[4], x5 = xp[5];
    const float4 g0 = gp[0], g1 = gp[1], g4 = gp[4], g5 = gp[5];
    const float hv[16] = {
        x0.x + g0.x, x0.y + g0.y, x0.z + g0.z, x0.w + g0.w,
        x1.x + g1.x, x1.y + g1.y, x1.z + g1.z, x1.w + g1.w,
        x4.x + g4.x, x4.y + g4.y, x4.z + g4.z, x4.w + g4.w,
        x5.x + g5.x, x5.y + g5.y, x5.z + g5.z, x5.w + g5.w };
    v16h a;
#pragma unroll
    for (int i = 0; i < 16; ++i) a[i] = (_Float16)hv[i];

    v8f c0 = {}, c1 = {};
    c0 = __builtin_amdgcn_wmma_f32_16x16x32_f16(false, a, false, bw1a, (short)0, c0, false, false);
    c1 = __builtin_amdgcn_wmma_f32_16x16x32_f16(false, a, false, bw1b, (short)0, c1, false, false);

    // bias + relu, stage y1 into LDS for the cross-lane transpose
    const float b1lo = B1[mr], b1hi = B1[16 + mr];
#pragma unroll
    for (int r = 0; r < 8; ++r) {
        Lp[(mb + r) * 33 + mr]      = fmaxf(c0[r] + b1lo, 0.f);
        Lp[(mb + r) * 33 + 16 + mr] = fmaxf(c1[r] + b1hi, 0.f);
    }
    __syncthreads();

    v16h a2;
#pragma unroll
    for (int i = 0; i < 16; ++i) a2[i] = (_Float16)Lp[mr * 33 + kmap(i, h)];

    v8f d0 = {}, d1 = {};
    d0 = __builtin_amdgcn_wmma_f32_16x16x32_f16(false, a2, false, bw2a, (short)0, d0, false, false);
    d1 = __builtin_amdgcn_wmma_f32_16x16x32_f16(false, a2, false, bw2b, (short)0, d1, false, false);

    const float b2lo = B2[mr], b2hi = B2[16 + mr];
    __syncthreads();
#pragma unroll
    for (int r = 0; r < 8; ++r) {
        Lp[(mb + r) * 33 + mr]      = fmaxf(d0[r] + b2lo, 0.f);
        Lp[(mb + r) * 33 + 16 + mr] = fmaxf(d1[r] + b2hi, 0.f);
    }
    __syncthreads();

    // coalesced z store + per-channel partial BN statistics (channel = lane)
    float s = 0.f, s2 = 0.f;
#pragma unroll
    for (int row = 0; row < 16; ++row) {
        const float v = Lp[row * 33 + lane];
        Z[(long)(node0 + row) * 32 + lane] = v;
        s += v; s2 += v * v;
    }
    ATOM_ADD(&stats[lane], s);
    ATOM_ADD(&stats[32 + lane], s2);
}

// -------------------- graph max-pool + linear (WMMA) -----------------------
__global__ void __launch_bounds__(256) k_pool_lin(
    const float* __restrict__ X, const float* __restrict__ WL,
    const float* __restrict__ BL, float* __restrict__ EB) {

    __shared__ float lds[8][16 * 33];
    const int lane = threadIdx.x & 31;
    const int wave = threadIdx.x >> 5;
    float* Lp = &lds[wave][0];
    const int g0 = (blockIdx.x * 8 + wave) * 16;
    const int mr = lane & 15;
    const int h  = lane >> 4;
    const int mb = h * 8;

    const v16h bl0 = make_bfrag(WL, 32, 0, lane);
    const v16h bl1 = make_bfrag(WL, 32, 16, lane);

    float acc[16];
#pragma unroll
    for (int i = 0; i < 16; ++i) acc[i] = -3.402823466e38f;
    const int g = g0 + mr;
    for (int j = 0; j < 8; ++j) {
        const float4* p = (const float4*)(X + (long)(8 * g + j) * 32 + h * 8);
        const float4 q0 = p[0], q1 = p[1], q4 = p[4], q5 = p[5];
        acc[0]  = fmaxf(acc[0],  q0.x); acc[1]  = fmaxf(acc[1],  q0.y);
        acc[2]  = fmaxf(acc[2],  q0.z); acc[3]  = fmaxf(acc[3],  q0.w);
        acc[4]  = fmaxf(acc[4],  q1.x); acc[5]  = fmaxf(acc[5],  q1.y);
        acc[6]  = fmaxf(acc[6],  q1.z); acc[7]  = fmaxf(acc[7],  q1.w);
        acc[8]  = fmaxf(acc[8],  q4.x); acc[9]  = fmaxf(acc[9],  q4.y);
        acc[10] = fmaxf(acc[10], q4.z); acc[11] = fmaxf(acc[11], q4.w);
        acc[12] = fmaxf(acc[12], q5.x); acc[13] = fmaxf(acc[13], q5.y);
        acc[14] = fmaxf(acc[14], q5.z); acc[15] = fmaxf(acc[15], q5.w);
    }
    v16h a;
#pragma unroll
    for (int i = 0; i < 16; ++i) a[i] = (_Float16)acc[i];

    v8f c0 = {}, c1 = {};
    c0 = __builtin_amdgcn_wmma_f32_16x16x32_f16(false, a, false, bl0, (short)0, c0, false, false);
    c1 = __builtin_amdgcn_wmma_f32_16x16x32_f16(false, a, false, bl1, (short)0, c1, false, false);

    const float blo = BL[mr], bhi = BL[16 + mr];
#pragma unroll
    for (int r = 0; r < 8; ++r) {
        Lp[(mb + r) * 33 + mr]      = fmaxf(c0[r] + blo, 0.f);
        Lp[(mb + r) * 33 + 16 + mr] = fmaxf(c1[r] + bhi, 0.f);
    }
    __syncthreads();
#pragma unroll
    for (int row = 0; row < 16; ++row)
        EB[(long)(g0 + row) * 32 + lane] = Lp[row * 33 + lane];
}

// ------------------------------ final head ---------------------------------
__global__ void __launch_bounds__(256) k_head(
    const float* __restrict__ EB, const float* __restrict__ ER,
    const float* __restrict__ Wb, const float* __restrict__ bb,
    const float* __restrict__ wm, const float* __restrict__ bm,
    float* __restrict__ out) {

    __shared__ float sW[64 * 16];
    __shared__ float sb[16];
    __shared__ float sm[16];
    __shared__ float sbm;
    const int t = threadIdx.x;
    for (int i = t; i < 1024; i += 256) sW[i] = Wb[i];
    if (t < 16) { sb[t] = bb[t]; sm[t] = wm[t]; }
    if (t == 0) sbm = bm[0];
    __syncthreads();

    const int gidx = blockIdx.x * 256 + t;   // < G
    float cat[64];
    const float4* pe = (const float4*)(EB + (long)gidx * 32);
    const float4* pr = (const float4*)(ER + (long)gidx * 32);
#pragma unroll
    for (int k = 0; k < 8; ++k) {
        const float4 v = pe[k];
        cat[4 * k + 0] = v.x; cat[4 * k + 1] = v.y;
        cat[4 * k + 2] = v.z; cat[4 * k + 3] = v.w;
    }
#pragma unroll
    for (int k = 0; k < 8; ++k) {
        const float4 v = pr[k];
        cat[32 + 4 * k + 0] = v.x; cat[32 + 4 * k + 1] = v.y;
        cat[32 + 4 * k + 2] = v.z; cat[32 + 4 * k + 3] = v.w;
    }
    float o = sbm;
    for (int j = 0; j < 16; ++j) {
        float hj = sb[j];
#pragma unroll
        for (int i = 0; i < 64; ++i) hj += cat[i] * sW[i * 16 + j];
        o += fmaxf(hj, 0.f) * sm[j];
    }
    out[gidx] = 1.0f / (1.0f + expf(-o));
}

// ------------------------------- launcher ----------------------------------
extern "C" void kernel_launch(void* const* d_in, const int* in_sizes, int n_in,
                              void* d_out, int out_size, void* d_ws, size_t ws_size,
                              hipStream_t stream) {
    (void)in_sizes; (void)n_in; (void)out_size; (void)ws_size;

    float* ws    = (float*)d_ws;
    float* X     = ws;
    float* AGG   = ws + (long)NF;
    float* Z     = ws + 2L * NF;
    float* EB    = ws + 3L * NF;
    float* ER    = EB + (long)G * 32;
    float* STATS = ER + (long)G * 32;     // 64 floats

    // d_in layout (flattened setup_inputs dict, leaf order):
    // 0 data_base 1 edge_index_base 2 batch_base 3 data_residual
    // 4 edge_index_residual 5 batch_residual
    // 6..9 conv1{W1,b1,W2,b2} 10..13 conv2 14..17 conv3
    // 18..21 conv1r 22..25 conv2r 26..29 conv3r
    // 30 bn_gamma[6,32] 31 bn_beta[6,32]
    // 32/33 lin_b1{W,b} 34/35 lin_b2{W,b} 36/37 lin_before{W,b} 38/39 lin_mean{W,b}

    for (int br = 0; br < 2; ++br) {
        const float* data = (const float*)d_in[br ? 3 : 0];
        const int*   ei   = (const int*)d_in[br ? 4 : 1];
        k_pack<<<NF / 256, 256, 0, stream>>>(data, X);
        for (int l = 0; l < 3; ++l) {
            const int wb = (br ? 18 : 6) + l * 4;
            k_zero<<<NF / 4 / 256, 256, 0, stream>>>(AGG, STATS);
            k_agg<<<E / 256, 256, 0, stream>>>(ei, X, AGG, l == 0 ? 8 : 32);
            k_mlp<<<N / 128, 256, 0, stream>>>(
                X, AGG,
                (const float*)d_in[wb],     (const float*)d_in[wb + 1],
                (const float*)d_in[wb + 2], (const float*)d_in[wb + 3],
                l == 0 ? 6 : 32, Z, STATS);
            const float* gam = (const float*)d_in[30] + (br * 3 + l) * 32;
            const float* bet = (const float*)d_in[31] + (br * 3 + l) * 32;
            k_bn<<<NF / 256, 256, 0, stream>>>(Z, STATS, gam, bet, X);
        }
        k_pool_lin<<<G / 128, 256, 0, stream>>>(
            X, (const float*)d_in[br ? 34 : 32], (const float*)d_in[br ? 35 : 33],
            br ? ER : EB);
    }
    k_head<<<G / 256, 256, 0, stream>>>(
        EB, ER, (const float*)d_in[36], (const float*)d_in[37],
        (const float*)d_in[38], (const float*)d_in[39], (float*)d_out);
}
